// Decoder_50672024159069
// MI455X (gfx1250) — matrix operand order repaired and checked by reading, not compiled, not run on hardware
//
#include <hip/hip_runtime.h>
#include <hip/hip_bf16.h>
#include <math.h>

typedef __attribute__((ext_vector_type(16))) __bf16   v16bf;
typedef __attribute__((ext_vector_type(8)))  float    v8f;
typedef __attribute__((ext_vector_type(8)))  unsigned v8u;

#define NEGV (-1e20f)

__device__ __forceinline__ __bf16 f2bf(float f) {
    union { float f; unsigned u; } a; a.f = f;
    unsigned r = a.u + 0x7fffu + ((a.u >> 16) & 1u);
    union { unsigned short s; __bf16 b; } o; o.s = (unsigned short)(r >> 16);
    return o.b;
}
__device__ __forceinline__ unsigned packbf2(float lo, float hi) {
    union { unsigned short s; __bf16 b; } l, h;
    l.b = f2bf(lo); h.b = f2bf(hi);
    return (unsigned)l.s | ((unsigned)h.s << 16);
}
__device__ __forceinline__ float sigf(float x) { return 1.0f / (1.0f + expf(-x)); }
__device__ __forceinline__ float lae(float a, float b) {
    float m = fmaxf(a, b);
    if (m <= -1e19f) return m;
    return m + log1pf(expf(-fabsf(a - b)));
}

// ---------------------------------------------------------------------------
// Pack W[N,K] (fp32, split across two sources along K) into WMMA B-layout,
// bf16: tile(nt,kt) -> lane L holds N=nt*16+L%16, dword v = K pair, table order.
// ---------------------------------------------------------------------------
__global__ void pack_b_kernel(const float* __restrict__ W0, int ld0,
                              const float* __restrict__ W1, int ld1,
                              int split, int N, int K, unsigned* __restrict__ out)
{
    int idx = blockIdx.x * blockDim.x + threadIdx.x;
    int KT = K >> 5;
    int total = (N >> 4) * KT * 32 * 8;
    if (idx >= total) return;
    int v    = idx & 7;
    int lane = (idx >> 3) & 31;
    int kt   = (idx >> 8) % KT;
    int nt   = (idx >> 8) / KT;
    int k = kt * 32 + ((v < 4) ? (((lane & 16) ? 8 : 0) + 2 * v) : (16 + ((lane & 16) ? 8 : 0) + 2 * (v - 4)));
    int n = nt * 16 + (lane & 15);
    float w0, w1;
    if (k < split) { w0 = W0[(size_t)n * ld0 + k];           w1 = W0[(size_t)n * ld0 + k + 1]; }
    else           { w0 = W1[(size_t)n * ld1 + (k - split)]; w1 = W1[(size_t)n * ld1 + (k - split) + 1]; }
    out[idx] = packbf2(w0, w1);
}

// ---------------------------------------------------------------------------
// Pack the per-step 32-row activation matrix (gathered from up to 3 logical
// segments along K) into WMMA A-layout bf16:
//  lane<16: m=mt*16+lane, elems 0..7 -> K kt*32+0..7, 8..15 -> +16..23
//  lane>=16: same row, runs shifted by 8.  dword v = consecutive k pair.
// Layout: [mt][kt][lane][8 dwords]
// ---------------------------------------------------------------------------
__global__ void pack_a_kernel(const float* __restrict__ s0, int ld0, int len0,
                              const float* __restrict__ s1, int ld1, int len1,
                              const float* __restrict__ s2, int ld2,
                              int KT, unsigned* __restrict__ out)
{
    int idx = blockIdx.x * blockDim.x + threadIdx.x;   // 2*KT*256
    if (idx >= 2 * KT * 256) return;
    int v    = idx & 7;
    int lane = (idx >> 3) & 31;
    int kt   = (idx >> 8) % KT;
    int m    = ((idx >> 8) / KT) * 16 + (lane & 15);
    int base = (lane & 16) ? 8 : 0;
    int koff = (v < 4) ? (base + 2 * v) : (16 + base + 2 * (v - 4));
    int k = kt * 32 + koff;

    float x0, x1;
    if (k < len0)             { const float* p = s0 + (size_t)m * ld0 + k;          x0 = p[0]; x1 = p[1]; }
    else if (k < len0 + len1) { const float* p = s1 + (size_t)m * ld1 + (k - len0); x0 = p[0]; x1 = p[1]; }
    else                      { const float* p = s2 + (size_t)m * ld2 + (k - len0 - len1); x0 = p[0]; x1 = p[1]; }
    out[idx] = packbf2(x0, x1);
}

// ---------------------------------------------------------------------------
// Per-step GEMM, both operands pre-packed bf16: C[32,N] = A @ W^T.
// Each wave: the single m-pair x 4 n-tiles; inner loop is pure b128+wmma.
// ---------------------------------------------------------------------------
__global__ void gemm_wmma_pp(const unsigned* __restrict__ Ap, const unsigned* __restrict__ Bp,
                             const float* __restrict__ bias, float* __restrict__ C,
                             int N, int KT)
{
    const int lane = threadIdx.x & 31;
    const int ng   = (blockIdx.x * blockDim.x + threadIdx.x) >> 5;   // n-group id
    if (ng * 64 >= N) return;

    v8f acc[2][4];
    for (int i = 0; i < 2; ++i) for (int j = 0; j < 4; ++j) acc[i][j] = (v8f)(0.0f);

    const unsigned* a0 = Ap + (size_t)lane * 8;                   // mt=0, stride 256/kt
    const unsigned* a1 = Ap + ((size_t)KT * 32 + lane) * 8;       // mt=1
    const unsigned* b0 = Bp + (((size_t)(ng * 4 + 0) * KT) * 32 + lane) * 8;
    const unsigned* b1 = Bp + (((size_t)(ng * 4 + 1) * KT) * 32 + lane) * 8;
    const unsigned* b2 = Bp + (((size_t)(ng * 4 + 2) * KT) * 32 + lane) * 8;
    const unsigned* b3 = Bp + (((size_t)(ng * 4 + 3) * KT) * 32 + lane) * 8;

    for (int kt = 0; kt < KT; ++kt) {
        v16bf aa0 = __builtin_bit_cast(v16bf, *(const v8u*)(a0 + (size_t)kt * 256));
        v16bf aa1 = __builtin_bit_cast(v16bf, *(const v8u*)(a1 + (size_t)kt * 256));
        v16bf bb;
        bb = __builtin_bit_cast(v16bf, *(const v8u*)(b0 + (size_t)kt * 256));
        acc[0][0] = __builtin_amdgcn_wmma_f32_16x16x32_bf16(false, aa0, false, bb, (short)0, acc[0][0], false, false);
        acc[1][0] = __builtin_amdgcn_wmma_f32_16x16x32_bf16(false, aa1, false, bb, (short)0, acc[1][0], false, false);
        bb = __builtin_bit_cast(v16bf, *(const v8u*)(b1 + (size_t)kt * 256));
        acc[0][1] = __builtin_amdgcn_wmma_f32_16x16x32_bf16(false, aa0, false, bb, (short)0, acc[0][1], false, false);
        acc[1][1] = __builtin_amdgcn_wmma_f32_16x16x32_bf16(false, aa1, false, bb, (short)0, acc[1][1], false, false);
        bb = __builtin_bit_cast(v16bf, *(const v8u*)(b2 + (size_t)kt * 256));
        acc[0][2] = __builtin_amdgcn_wmma_f32_16x16x32_bf16(false, aa0, false, bb, (short)0, acc[0][2], false, false);
        acc[1][2] = __builtin_amdgcn_wmma_f32_16x16x32_bf16(false, aa1, false, bb, (short)0, acc[1][2], false, false);
        bb = __builtin_bit_cast(v16bf, *(const v8u*)(b3 + (size_t)kt * 256));
        acc[0][3] = __builtin_amdgcn_wmma_f32_16x16x32_bf16(false, aa0, false, bb, (short)0, acc[0][3], false, false);
        acc[1][3] = __builtin_amdgcn_wmma_f32_16x16x32_bf16(false, aa1, false, bb, (short)0, acc[1][3], false, false);
    }
    for (int mt = 0; mt < 2; ++mt) {
        for (int nt = 0; nt < 4; ++nt) {
            int n = (ng * 4 + nt) * 16 + (lane & 15);
            int mbase = mt * 16 + ((lane & 16) ? 8 : 0);
            float bv = bias ? bias[n] : 0.0f;
            #pragma unroll
            for (int r = 0; r < 8; ++r)
                C[(size_t)(mbase + r) * N + n] = acc[mt][nt][r] + bv;
        }
    }
}

// ---------------------------------------------------------------------------
// Generic GEMM with fp32 A converted on the fly (used for the big parallel
// processed-memory GEMM, off the serial critical path).
// ---------------------------------------------------------------------------
__global__ void gemm_wmma(const float* __restrict__ X, const unsigned* __restrict__ Bp,
                          const float* __restrict__ bias, float* __restrict__ C,
                          int M, int N, int K)
{
    const int lane = threadIdx.x & 31;
    const int wave = (blockIdx.x * blockDim.x + threadIdx.x) >> 5;
    const int KT = K >> 5;
    const int ngroups = (N >> 4) >> 2;
    const int mpair = wave / ngroups;
    const int ng    = wave % ngroups;
    if (mpair * 32 >= M) return;

    v8f acc[2][4];
    for (int i = 0; i < 2; ++i) for (int j = 0; j < 4; ++j) acc[i][j] = (v8f)(0.0f);

    for (int kt = 0; kt < KT; ++kt) {
        v16bf a[2];
        #pragma unroll
        for (int mt = 0; mt < 2; ++mt) {
            int m = mpair * 32 + mt * 16 + (lane & 15);
            const float* xr = X + (size_t)m * K + kt * 32 + ((lane & 16) ? 8 : 0);
            #pragma unroll
            for (int e = 0; e < 8; ++e) { a[mt][e] = f2bf(xr[e]); a[mt][8 + e] = f2bf(xr[16 + e]); }
        }
        #pragma unroll
        for (int nt = 0; nt < 4; ++nt) {
            const unsigned* bp = Bp + (((size_t)(ng * 4 + nt) * KT + kt) * 32 + lane) * 8;
            v16bf bb = __builtin_bit_cast(v16bf, *(const v8u*)bp);
            acc[0][nt] = __builtin_amdgcn_wmma_f32_16x16x32_bf16(false, a[0], false, bb, (short)0, acc[0][nt], false, false);
            acc[1][nt] = __builtin_amdgcn_wmma_f32_16x16x32_bf16(false, a[1], false, bb, (short)0, acc[1][nt], false, false);
        }
    }
    for (int mt = 0; mt < 2; ++mt) {
        for (int nt = 0; nt < 4; ++nt) {
            int n = (ng * 4 + nt) * 16 + (lane & 15);
            int mbase = mpair * 32 + mt * 16 + ((lane & 16) ? 8 : 0);
            float bv = bias ? bias[n] : 0.0f;
            #pragma unroll
            for (int r = 0; r < 8; ++r)
                C[(size_t)(mbase + r) * N + n] = acc[mt][nt][r] + bv;
        }
    }
}

// ---------------------------------------------------------------------------
// Prenet (dense + relu + deterministic hash dropout, keep=0.5, inverted)
// ---------------------------------------------------------------------------
__device__ __forceinline__ float hashdrop(float x, unsigned idx, unsigned salt) {
    unsigned h = idx * 2654435761u ^ salt;
    h ^= h >> 13; h *= 0x85ebca6bu; h ^= h >> 16;
    return (h & 1u) ? x * 2.0f : 0.0f;
}

__global__ void prenet1_kernel(const float* __restrict__ dec_in, const float* __restrict__ w1,
                               float* __restrict__ out)
{
    int idx = blockIdx.x * blockDim.x + threadIdx.x;     // 500*32*256
    if (idx >= 500 * 32 * 256) return;
    int j = idx & 255, b = (idx >> 8) & 31, t = idx >> 13;
    float s = 0.0f;
    if (t > 0) {
        const float* x = dec_in + (size_t)b * 80 * 500 + (t - 1);  // [b, k, t-1]
        const float* w = w1 + j * 80;
        for (int k = 0; k < 80; ++k) s += x[(size_t)k * 500] * w[k];
    }
    s = fmaxf(s, 0.0f);
    out[idx] = hashdrop(s, (unsigned)idx, 0xA5A5A5A5u);
}

__global__ void prenet2_kernel(const float* __restrict__ p1, const float* __restrict__ w2,
                               float* __restrict__ out)
{
    int idx = blockIdx.x * blockDim.x + threadIdx.x;
    if (idx >= 500 * 32 * 256) return;
    int j = idx & 255;
    const float* x = p1 + (size_t)(idx >> 8) * 256;
    const float* w = w2 + j * 256;
    float s = 0.0f;
    for (int k = 0; k < 256; ++k) s += x[k] * w[k];
    s = fmaxf(s, 0.0f);
    out[idx] = hashdrop(s, (unsigned)idx, 0x5A5A5A5Au);
}

// ---------------------------------------------------------------------------
// LSTM gate nonlinearity + zoneout
// ---------------------------------------------------------------------------
__global__ void lstm_pw_kernel(const float* __restrict__ gates, float* __restrict__ h,
                               float* __restrict__ c)
{
    int idx = blockIdx.x * blockDim.x + threadIdx.x;     // 32*1024
    if (idx >= 32 * 1024) return;
    int j = idx & 1023, b = idx >> 10;
    const float* g = gates + (size_t)b * 4096;
    float ig = sigf(g[j]), fg = sigf(g[1024 + j]);
    float gg = tanhf(g[2048 + j]), og = sigf(g[3072 + j]);
    float c2 = fg * c[idx] + ig * gg;
    float h2 = og * tanhf(c2);
    h[idx] = 0.1f * h[idx] + 0.9f * h2;
    c[idx] = 0.1f * c[idx] + 0.9f * c2;
}

// ---------------------------------------------------------------------------
// Location-sensitive forward attention: one workgroup per batch element
// ---------------------------------------------------------------------------
__global__ void attention_kernel(const float* __restrict__ ah, const float* __restrict__ wq,
                                 const float* __restrict__ vvec, const float* __restrict__ loc_conv,
                                 const float* __restrict__ loc_dense, const float* __restrict__ pm,
                                 const float* __restrict__ memory,
                                 float* __restrict__ aw_buf, float* __restrict__ w_cum,
                                 float* __restrict__ log_alpha, float* __restrict__ ctx,
                                 float* __restrict__ out_align, int t)
{
    __shared__ float s_pq[128];
    __shared__ float s_wp[232], s_wc[232];
    __shared__ float s_la[200], s_aw[200];
    __shared__ float s_red[256];
    int b = blockIdx.x, tid = threadIdx.x;

    for (int j = tid; j < 128; j += 256) {
        const float* w = wq + (size_t)j * 1024;
        const float* h = ah + (size_t)b * 1024;
        float s = 0.0f;
        for (int k = 0; k < 1024; ++k) s += h[k] * w[k];
        s_pq[j] = s;
    }
    for (int i = tid; i < 230; i += 256) {
        int t2 = i - 15;
        bool in = (t2 >= 0 && t2 < 200);
        s_wp[i] = in ? aw_buf[b * 200 + t2] : 0.0f;
        s_wc[i] = in ? w_cum[b * 200 + t2] : 0.0f;
    }
    __syncthreads();

    for (int t2 = tid; t2 < 200; t2 += 256) {
        float f[32];
        for (int c = 0; c < 32; ++c) {
            const float* cw = loc_conv + c * 62;
            float s = 0.0f;
            for (int k = 0; k < 31; ++k) s += s_wp[t2 + k] * cw[k] + s_wc[t2 + k] * cw[31 + k];
            f[c] = s;
        }
        float e = 0.0f;
        const float* pmr = pm + ((size_t)b * 200 + t2) * 128;
        for (int j = 0; j < 128; ++j) {
            const float* ld = loc_dense + j * 32;
            float pa = 0.0f;
            for (int c = 0; c < 32; ++c) pa += ld[c] * f[c];
            e += vvec[j] * tanhf(s_pq[j] + pa + pmr[j]);
        }
        float la_t = log_alpha[b * 200 + t2];
        float la_p = (t2 > 0) ? log_alpha[b * 200 + t2 - 1] : NEGV;
        s_la[t2] = lae(la_t, la_p) + e;
    }
    __syncthreads();

    float m = -3.0e38f;
    for (int t2 = tid; t2 < 200; t2 += 256) m = fmaxf(m, s_la[t2]);
    s_red[tid] = m; __syncthreads();
    for (int s = 128; s > 0; s >>= 1) { if (tid < s) s_red[tid] = fmaxf(s_red[tid], s_red[tid + s]); __syncthreads(); }
    m = s_red[0]; __syncthreads();

    float sum = 0.0f;
    for (int t2 = tid; t2 < 200; t2 += 256) { float e = expf(s_la[t2] - m); s_aw[t2] = e; sum += e; }
    s_red[tid] = sum; __syncthreads();
    for (int s = 128; s > 0; s >>= 1) { if (tid < s) s_red[tid] += s_red[tid + s]; __syncthreads(); }
    float inv = 1.0f / s_red[0];

    for (int t2 = tid; t2 < 200; t2 += 256) {
        float a = s_aw[t2] * inv; s_aw[t2] = a;
        aw_buf[b * 200 + t2] = a;
        w_cum[b * 200 + t2] += a;
        log_alpha[b * 200 + t2] = s_la[t2];
        out_align[((size_t)b * 500 + t) * 200 + t2] = a;
    }
    __syncthreads();

    for (int e = tid; e < 832; e += 256) {
        const float* mem = memory + (size_t)b * 200 * 832 + e;
        float s = 0.0f;
        for (int t2 = 0; t2 < 200; ++t2) s += s_aw[t2] * mem[(size_t)t2 * 832];
        ctx[b * 832 + e] = s;
    }
}

// ---------------------------------------------------------------------------
// Mel projection + stop gate
// ---------------------------------------------------------------------------
__global__ void proj_kernel(const float* __restrict__ dh, const float* __restrict__ ctx,
                            const float* __restrict__ proj_w, const float* __restrict__ proj_b,
                            const float* __restrict__ gate_w, const float* __restrict__ gate_b,
                            float* __restrict__ out, int t)
{
    int idx = blockIdx.x * blockDim.x + threadIdx.x;     // 32*81
    if (idx >= 32 * 81) return;
    int b = idx / 81, r = idx % 81;
    const float* w = (r < 80) ? proj_w + (size_t)r * 1856 : gate_w;
    float s = (r < 80) ? proj_b[r] : gate_b[0];
    const float* h = dh + (size_t)b * 1024;
    const float* c = ctx + (size_t)b * 832;
    for (int k = 0; k < 1024; ++k) s += h[k] * w[k];
    for (int k = 0; k < 832; ++k)  s += c[k] * w[1024 + k];
    if (r < 80) out[((size_t)b * 80 + r) * 500 + t] = s;
    else        out[1280000 + (size_t)b * 500 + t] = s;
}

__global__ void fillzero_kernel(float* __restrict__ p, int n) {
    int idx = blockIdx.x * blockDim.x + threadIdx.x;
    if (idx < n) p[idx] = 0.0f;
}
__global__ void init_logalpha_kernel(float* __restrict__ la) {
    int idx = blockIdx.x * blockDim.x + threadIdx.x;
    if (idx < 32 * 200) la[idx] = ((idx % 200) == 0) ? 0.0f : NEGV;
}

// ---------------------------------------------------------------------------
extern "C" void kernel_launch(void* const* d_in, const int* in_sizes, int n_in,
                              void* d_out, int out_size, void* d_ws, size_t ws_size,
                              hipStream_t stream)
{
    (void)in_sizes; (void)n_in; (void)out_size; (void)ws_size;
    const float* memory    = (const float*)d_in[0];
    const float* dec_in    = (const float*)d_in[1];
    const float* pre_w1    = (const float*)d_in[2];
    const float* pre_w2    = (const float*)d_in[3];
    const float* arnn_wih  = (const float*)d_in[4];
    const float* arnn_whh  = (const float*)d_in[5];
    const float* arnn_b    = (const float*)d_in[6];
    const float* wq        = (const float*)d_in[7];
    const float* wm        = (const float*)d_in[8];
    const float* vvec      = (const float*)d_in[9];
    const float* loc_conv  = (const float*)d_in[10];
    const float* loc_dense = (const float*)d_in[11];
    const float* drnn_wih  = (const float*)d_in[12];
    const float* drnn_whh  = (const float*)d_in[13];
    const float* drnn_b    = (const float*)d_in[14];
    const float* proj_w    = (const float*)d_in[15];
    const float* proj_b    = (const float*)d_in[16];
    const float* gate_w    = (const float*)d_in[17];
    const float* gate_b    = (const float*)d_in[18];
    float* out = (float*)d_out;

    const int KTa = 2112 / 32;   // 66
    const int KTd = 2880 / 32;   // 90

    // ---- workspace layout (4-byte units) ----
    float* ws = (float*)d_ws;
    size_t off = 0;
    auto take = [&](size_t n) { size_t o = off; off += n; return o; };
    size_t o_pm    = take(6400u * 128);              // processed memory
    size_t o_p1    = take(500u * 32 * 256);          // prenet layer 1
    size_t o_p2    = take(500u * 32 * 256);          // prenet layer 2
    size_t o_pWa   = take((size_t)4096 * 2112 / 2);  // packed bf16 weights (dwords)
    size_t o_pWd   = take((size_t)4096 * 2880 / 2);
    size_t o_pWm   = take((size_t)128 * 832 / 2);
    size_t o_pAa   = take((size_t)2 * KTa * 256);    // packed A, attn step
    size_t o_pAd   = take((size_t)2 * KTd * 256);    // packed A, dec step
    size_t o_ga    = take(32u * 4096);               // attn gates
    size_t o_gd    = take(32u * 4096);               // dec gates
    size_t o_state = off;                            // ---- zeroed block ----
    size_t o_ah    = take(32u * 1024);
    size_t o_ac    = take(32u * 1024);
    size_t o_dh    = take(32u * 1024);
    size_t o_dc    = take(32u * 1024);
    size_t o_ctx   = take(32u * 832);
    size_t o_aw    = take(32u * 200);
    size_t o_wcum  = take(32u * 200);
    size_t n_state = off - o_state;                  // ---- end zeroed ----
    size_t o_la    = take(32u * 200);

    float*    pm      = ws + o_pm;
    float*    p1      = ws + o_p1;
    float*    p2      = ws + o_p2;
    unsigned* pWa     = (unsigned*)(ws + o_pWa);
    unsigned* pWd     = (unsigned*)(ws + o_pWd);
    unsigned* pWm     = (unsigned*)(ws + o_pWm);
    unsigned* pAa     = (unsigned*)(ws + o_pAa);
    unsigned* pAd     = (unsigned*)(ws + o_pAd);
    float*    gates_a = ws + o_ga;
    float*    gates_d = ws + o_gd;
    float*    ah      = ws + o_ah;
    float*    ac      = ws + o_ac;
    float*    dh      = ws + o_dh;
    float*    dc      = ws + o_dc;
    float*    ctx     = ws + o_ctx;
    float*    aw_buf  = ws + o_aw;
    float*    w_cum   = ws + o_wcum;
    float*    la      = ws + o_la;

    // ---- one-time (per call) setup ----
    {
        int thrA = (4096 / 16) * KTa * 256;
        int thrD = (4096 / 16) * KTd * 256;
        int thrM = (128 / 16) * (832 / 32) * 256;
        pack_b_kernel<<<(thrA + 255) / 256, 256, 0, stream>>>(arnn_wih, 1088, arnn_whh, 1024, 1088, 4096, 2112, pWa);
        pack_b_kernel<<<(thrD + 255) / 256, 256, 0, stream>>>(drnn_wih, 1856, drnn_whh, 1024, 1856, 4096, 2880, pWd);
        pack_b_kernel<<<(thrM + 255) / 256, 256, 0, stream>>>(wm, 832, (const float*)nullptr, 0, 832, 128, 832, pWm);

        fillzero_kernel<<<((int)n_state + 255) / 256, 256, 0, stream>>>(ws + o_state, (int)n_state);
        init_logalpha_kernel<<<(6400 + 255) / 256, 256, 0, stream>>>(la);

        prenet1_kernel<<<16000, 256, 0, stream>>>(dec_in, pre_w1, p1);
        prenet2_kernel<<<16000, 256, 0, stream>>>(p1, pre_w2, p2);

        // pm = memory @ wm^T : M=6400, N=128, K=832 -> 400 waves
        gemm_wmma<<<(400 * 32 + 255) / 256, 256, 0, stream>>>(memory, pWm, nullptr, pm, 6400, 128, 832);
    }

    float* out_align = out + 1296000;  // mel 1,280,000 + gate 16,000

    // ---- 500 sequential decode steps ----
    const int gA = (2 * KTa * 256 + 255) / 256;
    const int gD = (2 * KTd * 256 + 255) / 256;
    for (int t = 0; t < 500; ++t) {
        // A = [pren2(t) | ctx | ah]  (all current from previous step / setup)
        pack_a_kernel<<<gA, 256, 0, stream>>>(p2 + (size_t)t * 32 * 256, 256, 256,
                                              ctx, 832, 832, ah, 1024, KTa, pAa);
        gemm_wmma_pp<<<8, 256, 0, stream>>>(pAa, pWa, arnn_b, gates_a, 4096, KTa);
        lstm_pw_kernel<<<128, 256, 0, stream>>>(gates_a, ah, ac);
        attention_kernel<<<32, 256, 0, stream>>>(ah, wq, vvec, loc_conv, loc_dense, pm, memory,
                                                 aw_buf, w_cum, la, ctx, out_align, t);
        // A = [ah | ctx | dh_prev]
        pack_a_kernel<<<gD, 256, 0, stream>>>(ah, 1024, 1024, ctx, 832, 832, dh, 1024, KTd, pAd);
        gemm_wmma_pp<<<8, 256, 0, stream>>>(pAd, pWd, drnn_b, gates_d, 4096, KTd);
        lstm_pw_kernel<<<128, 256, 0, stream>>>(gates_d, dh, dc);
        proj_kernel<<<(32 * 81 + 255) / 256, 256, 0, stream>>>(dh, ctx, proj_w, proj_b,
                                                               gate_w, gate_b, out, t);
    }
}
